// DownBlock_30227979829916
// MI455X (gfx1250) — compile-verified, matrix operand order checked by reading, not run on hardware
//
#include <hip/hip_runtime.h>

// ---------------------------------------------------------------------------
// CDNA5 (gfx1250) types for V_WMMA_F32_16X16X32_BF16
// ---------------------------------------------------------------------------
typedef __bf16 bf16;
typedef bf16 v16bf __attribute__((ext_vector_type(16)));
typedef bf16 v8bf  __attribute__((ext_vector_type(8)));
typedef float v8f  __attribute__((ext_vector_type(8)));
typedef unsigned int u32x4 __attribute__((ext_vector_type(4)));
typedef unsigned int u32x8 __attribute__((ext_vector_type(8)));

union Frag { v16bf v; v8bf h[2]; };

__device__ __forceinline__ unsigned short f2bf(float f) {
    unsigned int u = __float_as_uint(f);
    u += 0x7fffu + ((u >> 16) & 1u);      // round-to-nearest-even
    return (unsigned short)(u >> 16);
}

// ---------------------------------------------------------------------------
// Tensor Data Mover: 2D tile load (bf16 elements) global -> LDS.
// Descriptor per CDNA5 ISA (D# groups 0/1; groups 2/3
// zero for 2D tiles).  tensor dims == tile dims so no OOB path is taken.
// Issued by a single wave; completion tracked with TENSORcnt.
// ---------------------------------------------------------------------------
__device__ __forceinline__ void tdm_load_2d_bf16(unsigned lds_addr, const void* gptr,
                                                 unsigned tile_w, unsigned tile_h,
                                                 unsigned stride_elems) {
    unsigned long long ga = (unsigned long long)gptr;
    u32x4 g0;
    g0[0] = 1u;                                         // count=1, user desc
    g0[1] = lds_addr;                                   // LDS byte address
    g0[2] = (unsigned)(ga & 0xffffffffu);               // global_addr[31:0]
    g0[3] = (unsigned)((ga >> 32) & 0x1ffffffu)         // global_addr[56:32]
          | (2u << 30);                                 // type = 2 ("image")
    u32x8 g1;
    g1[0] = 1u << 16;                                   // data_size = 1 (2 bytes)
    g1[1] = (tile_w & 0xffffu) << 16;                   // tensor_dim0[15:0]
    g1[2] = ((tile_w >> 16) & 0xffffu)                  // tensor_dim0[31:16]
          | ((tile_h & 0xffffu) << 16);                 // tensor_dim1[15:0]
    g1[3] = ((tile_h >> 16) & 0xffffu)                  // tensor_dim1[31:16]
          | ((tile_w & 0xffffu) << 16);                 // tile_dim0
    g1[4] = tile_h & 0xffffu;                           // tile_dim1 (tile_dim2=0)
    g1[5] = stride_elems;                               // tensor_dim0_stride[31:0]
    g1[6] = 0u;                                         // stride[47:32], dim1_stride lo
    g1[7] = 0u;
    u32x4 z4 = {0u, 0u, 0u, 0u};
    asm volatile("tensor_load_to_lds %0, %1, %2, %3"
                 :: "s"(g0), "s"(g1), "s"(z4), "s"(z4)
                 : "memory");
}

__device__ __forceinline__ unsigned lds_offset(const void* p) {
    // generic pointers into the LDS aperture carry the LDS byte offset in [31:0]
    return (unsigned)(unsigned long long)p;
}

// ---------------------------------------------------------------------------
// flat f32 -> bf16 cast
// ---------------------------------------------------------------------------
__global__ void cast_bf16_k(const float* __restrict__ s, unsigned short* __restrict__ d,
                            long long n) {
    long long i = (long long)blockIdx.x * blockDim.x + threadIdx.x;
    if (i < n) d[i] = f2bf(s[i]);
}

// [4][C][4096] f32 -> [4*4096][C] bf16 (token-major, coalesced writes)
template <int C>
__global__ void transpose_cast_k(const float* __restrict__ s, unsigned short* __restrict__ d) {
    constexpr int NPB = 4096;
    long long total = (long long)4 * C * NPB;
    long long i = (long long)blockIdx.x * blockDim.x + threadIdx.x;
    if (i >= total) return;
    int c = (int)(i % C);                 // C is a power of two -> mask
    long long n = i / C;
    int b = (int)(n >> 12);
    int p = (int)(n & (NPB - 1));
    d[i] = f2bf(s[((long long)b * C + c) * NPB + p]);
}

// 3x3 SAME im2col: src [4][C][64][64] f32 -> dst [4*4096][C*9] bf16
template <int C>
__global__ void im2col3x3_k(const float* __restrict__ s, unsigned short* __restrict__ d) {
    constexpr int H = 64, W = 64, HW = H * W, K = C * 9;
    long long total = (long long)4 * HW * K;
    long long i = (long long)blockIdx.x * blockDim.x + threadIdx.x;
    if (i >= total) return;
    int kk = (int)(i % K);                // constant divisor -> mul/shift
    long long n = i / K;
    int c = kk / 9, r = kk % 9, ky = r / 3, kx = r % 3;
    int b = (int)(n >> 12), p = (int)(n & (HW - 1));
    int y = p >> 6, x0 = p & 63;
    int yy = y + ky - 1, xx = x0 + kx - 1;
    float v = 0.f;
    if (yy >= 0 && yy < H && xx >= 0 && xx < W)
        v = s[(((long long)b * C + c) * H + yy) * W + xx];
    d[i] = f2bf(v);
}

// ---------------------------------------------------------------------------
// TDM-fed WMMA GEMM:  out[M][16384] = A[M][K](bf16) * Bm[16384][K]^T(bf16)
// block = 256 threads (8 waves); grid = (M/32, 16384/128)
// Block tile 32(M) x 128(N).  Wave w owns N sub-tile w*16; each wave does
// 2 WMMAs per K-step (M sub-tiles 0/16) reusing its B fragment.
// A (32x32) and B (128x32) K-panels are DMA'd into double-buffered LDS by
// wave 0 via tensor_load_to_lds, overlapped with WMMA on the other buffer.
// layout 0: out[((n>>12)*M + m)*4096 + (n&4095)]   ([4][M][4096])
// layout 1: out[n*M + m]                           ([16384][M], token-major)
// ---------------------------------------------------------------------------
__global__ __launch_bounds__(256)
void wmma_gemm_k(const unsigned short* __restrict__ A, const unsigned short* __restrict__ Bm,
                 const float* __restrict__ bias, void* __restrict__ out,
                 int M, int K, int layout, int out_bf16) {
    __shared__ unsigned short sA[2][32 * 32];    // 2 x 2KB
    __shared__ unsigned short sB[2][128 * 32];   // 2 x 8KB
    int lane = threadIdx.x & 31;
    int wave = threadIdx.x >> 5;
    int m0 = blockIdx.x * 32;
    int n0 = blockIdx.y * 128;
    int row  = lane & 15;
    int koff = (lane >> 4) * 8;

    unsigned ldsA = lds_offset(&sA[0][0]);
    unsigned ldsB = lds_offset(&sB[0][0]);

    v8f acc0 = {}, acc1 = {};
    bool loader = (wave == 0);

    if (loader) {
        tdm_load_2d_bf16(ldsA, A + (long long)m0 * K, 32, 32, (unsigned)K);
        tdm_load_2d_bf16(ldsB, Bm + (long long)n0 * K, 32, 128, (unsigned)K);
    }
    __builtin_amdgcn_s_wait_tensorcnt(0);        // no-op for non-loader waves
    __syncthreads();

    for (int k = 0; k < K; k += 32) {
        int buf = (k >> 5) & 1;
        if (loader && (k + 32) < K) {            // prefetch next K-panel via TDM
            int nb = buf ^ 1;
            tdm_load_2d_bf16(ldsA + nb * (32 * 32 * 2),
                             A + (long long)m0 * K + (k + 32), 32, 32, (unsigned)K);
            tdm_load_2d_bf16(ldsB + nb * (128 * 32 * 2),
                             Bm + (long long)n0 * K + (k + 32), 32, 128, (unsigned)K);
        }

        Frag a0, a1, b;
        const unsigned short* pa = &sA[buf][0];
        a0.h[0] = *(const v8bf*)&pa[row * 32 + koff];
        a0.h[1] = *(const v8bf*)&pa[row * 32 + koff + 16];
        a1.h[0] = *(const v8bf*)&pa[(16 + row) * 32 + koff];
        a1.h[1] = *(const v8bf*)&pa[(16 + row) * 32 + koff + 16];
        const unsigned short* pb = &sB[buf][(wave * 16 + row) * 32];
        b.h[0] = *(const v8bf*)&pb[koff];
        b.h[1] = *(const v8bf*)&pb[koff + 16];

        acc0 = __builtin_amdgcn_wmma_f32_16x16x32_bf16(false, a0.v, false, b.v,
                                                       (short)0, acc0, false, false);
        acc1 = __builtin_amdgcn_wmma_f32_16x16x32_bf16(false, a1.v, false, b.v,
                                                       (short)0, acc1, false, false);

        __builtin_amdgcn_s_wait_tensorcnt(0);    // unconditional: 0 for non-loaders
        __syncthreads();
    }

#pragma unroll
    for (int s = 0; s < 2; ++s) {
        v8f acc = s ? acc1 : acc0;
#pragma unroll
        for (int r = 0; r < 8; ++r) {
            int m = m0 + s * 16 + r + ((lane >> 4) << 3);
            int n = n0 + wave * 16 + (lane & 15);
            float v = acc[r];
            if (bias) v += bias[m];
            long long idx = layout ? ((long long)n * M + m)
                                   : (((long long)(n >> 12) * M + m) * 4096ll + (n & 4095));
            if (out_bf16) ((unsigned short*)out)[idx] = f2bf(v);
            else          ((float*)out)[idx] = v;
        }
    }
}

// ---------------------------------------------------------------------------
// In-place GroupNorm + optional ReLU + optional time-embedding add.
// C=256, npb=4096 fixed.  grid = 4*groups, block = 256
// ---------------------------------------------------------------------------
__global__ __launch_bounds__(256)
void group_norm_k(float* __restrict__ data, const float* __restrict__ g,
                  const float* __restrict__ bta, const float* __restrict__ temb,
                  int relu, int groups) {
    constexpr int C = 256, NPB = 4096;
    __shared__ float s1[256], s2[256];
    int b   = blockIdx.x / groups;
    int grp = blockIdx.x % groups;
    int cpg = C / groups;
    long long base = ((long long)b * C + (long long)grp * cpg) * NPB;
    long long cnt  = (long long)cpg * NPB;

    float a = 0.f, q = 0.f;
    for (long long i = threadIdx.x; i < cnt; i += 256) {
        float v = data[base + i];
        a += v; q += v * v;
    }
    s1[threadIdx.x] = a; s2[threadIdx.x] = q;
    __syncthreads();
    for (int o = 128; o > 0; o >>= 1) {
        if ((int)threadIdx.x < o) {
            s1[threadIdx.x] += s1[threadIdx.x + o];
            s2[threadIdx.x] += s2[threadIdx.x + o];
        }
        __syncthreads();
    }
    float mean = s1[0] / (float)cnt;
    float var  = s2[0] / (float)cnt - mean * mean;
    float inv  = rsqrtf(var + 1e-5f);

    for (long long i = threadIdx.x; i < cnt; i += 256) {
        int c = grp * cpg + (int)(i >> 12);
        float v = (data[base + i] - mean) * inv * g[c] + bta[c];
        if (relu) v = fmaxf(v, 0.f);
        if (temb) v += temb[b * C + c];
        data[base + i] = v;
    }
}

// temb[b][m] = t[b] . W[m] + bias[m]   (tiny: 4x256, K=512)
__global__ void time_emb_k(const float* __restrict__ t, const float* __restrict__ w,
                           const float* __restrict__ bb, float* __restrict__ o,
                           int Kt, int M) {
    int m = threadIdx.x, b = blockIdx.x;
    if (m >= M) return;
    float s = bb[m];
    for (int k = 0; k < Kt; ++k) s += t[b * Kt + k] * w[m * Kt + k];
    o[b * M + m] = s;
}

__global__ void add_silu_k(const float* __restrict__ a, const float* __restrict__ b,
                           float* __restrict__ o, long long n) {
    long long i = (long long)blockIdx.x * blockDim.x + threadIdx.x;
    if (i < n) {
        float v = a[i] + b[i];
        o[i] = v / (1.f + __expf(-v));
    }
}

// ---------------------------------------------------------------------------
// Flash attention, one wave (32 threads) per 16-token tile.  B=4, N=4096, C=256.
// q,k: [B][N][32] bf16 ; v: [B][256][N] bf16 ; out = gamma*softmax(qk)v + xres
// Scores: one WMMA per 16x16 tile (K = 32 channels exactly).
// P (C-layout) -> A-layout via 1KB LDS bounce; O: 16 accum tiles (16x256).
// ---------------------------------------------------------------------------
__global__ __launch_bounds__(32)
void flash_attn_k(const unsigned short* __restrict__ q, const unsigned short* __restrict__ kk,
                  const unsigned short* __restrict__ vv, const float* __restrict__ xres,
                  float* __restrict__ out, const float* __restrict__ gp) {
    constexpr int NPB = 4096, C = 256;
    __shared__ unsigned short pbuf[16 * 32];
    int b  = blockIdx.x >> 8;                 // 256 tiles per batch
    int n0 = (blockIdx.x & 255) * 16;
    int lane = threadIdx.x;
    int row  = lane & 15;
    int koff = (lane >> 4) * 8;
    int half = lane >> 4;

    Frag aq;
    const unsigned short* qp = q + ((long long)b * NPB + n0 + row) * 32;
    aq.h[0] = *(const v8bf*)(qp + koff);
    aq.h[1] = *(const v8bf*)(qp + koff + 16);

    float mrow[8], lrow[8];
#pragma unroll
    for (int r = 0; r < 8; ++r) { mrow[r] = -1e30f; lrow[r] = 0.f; }
    v8f o16[16];
#pragma unroll
    for (int t = 0; t < 16; ++t) o16[t] = {};

    for (int m0 = 0; m0 < NPB; m0 += 32) {
        v8f s0 = {}, s1v = {};
        {
            Frag bk;
            const unsigned short* kp = kk + ((long long)b * NPB + m0 + row) * 32;
            bk.h[0] = *(const v8bf*)(kp + koff);
            bk.h[1] = *(const v8bf*)(kp + koff + 16);
            s0 = __builtin_amdgcn_wmma_f32_16x16x32_bf16(false, aq.v, false, bk.v,
                                                         (short)0, s0, false, false);
            const unsigned short* kp2 = kp + 16 * 32;
            bk.h[0] = *(const v8bf*)(kp2 + koff);
            bk.h[1] = *(const v8bf*)(kp2 + koff + 16);
            s1v = __builtin_amdgcn_wmma_f32_16x16x32_bf16(false, aq.v, false, bk.v,
                                                          (short)0, s1v, false, false);
        }
#pragma unroll
        for (int r = 0; r < 8; ++r) {
            float mx = fmaxf(s0[r], s1v[r]);
            for (int off = 1; off < 16; off <<= 1) mx = fmaxf(mx, __shfl_xor(mx, off, 32));
            float nm = fmaxf(mrow[r], mx);
            float sc = __expf(mrow[r] - nm);
            float p0 = __expf(s0[r] - nm);
            float p1 = __expf(s1v[r] - nm);
            float rs = p0 + p1;
            for (int off = 1; off < 16; off <<= 1) rs += __shfl_xor(rs, off, 32);
            lrow[r] = lrow[r] * sc + rs;
            mrow[r] = nm;
#pragma unroll
            for (int t = 0; t < 16; ++t) o16[t][r] *= sc;
            int n = r + (half << 3);
            pbuf[n * 32 + (lane & 15)]      = f2bf(p0);
            pbuf[n * 32 + 16 + (lane & 15)] = f2bf(p1);
        }
        __syncthreads();
        Frag ap;
        ap.h[0] = *(const v8bf*)&pbuf[row * 32 + koff];
        ap.h[1] = *(const v8bf*)&pbuf[row * 32 + koff + 16];
#pragma unroll
        for (int t = 0; t < 16; ++t) {
            Frag bv;
            const unsigned short* vp =
                vv + ((long long)b * C + t * 16 + (lane & 15)) * NPB + m0 + koff;
            bv.h[0] = *(const v8bf*)vp;
            bv.h[1] = *(const v8bf*)(vp + 16);
            o16[t] = __builtin_amdgcn_wmma_f32_16x16x32_bf16(false, ap.v, false, bv.v,
                                                             (short)0, o16[t], false, false);
        }
        __syncthreads();
    }

    float gamma = gp[0];
#pragma unroll
    for (int t = 0; t < 16; ++t) {
#pragma unroll
        for (int r = 0; r < 8; ++r) {
            int n = n0 + r + (half << 3);
            int c = t * 16 + (lane & 15);
            long long idx = ((long long)b * C + c) * NPB + n;
            out[idx] = gamma * (o16[t][r] / lrow[r]) + xres[idx];
        }
    }
}

// 2x2 max pool: [4][256][64][64] -> [4][256][32][32]
__global__ void maxpool_k(const float* __restrict__ s, float* __restrict__ d) {
    constexpr int C = 256, H = 64, W = 64, Ho = 32, Wo = 32;
    long long total = (long long)4 * C * Ho * Wo;
    long long i = (long long)blockIdx.x * blockDim.x + threadIdx.x;
    if (i >= total) return;
    int xo = (int)(i & (Wo - 1));
    int yo = (int)((i >> 5) & (Ho - 1));
    long long bc = i >> 10;                 // b*C + c
    const float* p = s + (bc * H + yo * 2) * W + xo * 2;
    d[i] = fmaxf(fmaxf(p[0], p[1]), fmaxf(p[W], p[W + 1]));
}

// ---------------------------------------------------------------------------
extern "C" void kernel_launch(void* const* d_in, const int* in_sizes, int n_in,
                              void* d_out, int out_size, void* d_ws, size_t ws_size,
                              hipStream_t stream) {
    const float* x      = (const float*)d_in[0];
    const float* t      = (const float*)d_in[1];
    const float* r1c1w  = (const float*)d_in[2];
    const float* r1tw   = (const float*)d_in[3];
    const float* r1tb   = (const float*)d_in[4];
    const float* r1g1g  = (const float*)d_in[5];
    const float* r1g1b  = (const float*)d_in[6];
    const float* r1c2w  = (const float*)d_in[7];
    const float* r1g2g  = (const float*)d_in[8];
    const float* r1g2b  = (const float*)d_in[9];
    const float* r1scw  = (const float*)d_in[10];
    const float* r1scg  = (const float*)d_in[11];
    const float* r1scb  = (const float*)d_in[12];
    const float* r2c1w  = (const float*)d_in[13];
    const float* r2tw   = (const float*)d_in[14];
    const float* r2tb   = (const float*)d_in[15];
    const float* r2g1g  = (const float*)d_in[16];
    const float* r2g1b  = (const float*)d_in[17];
    const float* r2c2w  = (const float*)d_in[18];
    const float* r2g2g  = (const float*)d_in[19];
    const float* r2g2b  = (const float*)d_in[20];
    const float* aqw    = (const float*)d_in[21];
    const float* aqb    = (const float*)d_in[22];
    const float* akw    = (const float*)d_in[23];
    const float* akb    = (const float*)d_in[24];
    const float* avw    = (const float*)d_in[25];
    const float* avb    = (const float*)d_in[26];
    const float* agam   = (const float*)d_in[27];

    const int B = 4, NPB = 4096;
    const int C0 = 128, C1 = 256;
    const long long NT = (long long)B * NPB;        // 16384

    char* ws = (char*)d_ws;
    size_t off = 0;
    auto alloc = [&](size_t bytes) -> void* {
        void* p = ws + off;
        off += (bytes + 255) & ~(size_t)255;
        return p;
    };
    unsigned short* ic  = (unsigned short*)alloc((size_t)NT * 2304 * 2);   // im2col
    unsigned short* xt  = (unsigned short*)alloc((size_t)NT * 256 * 2);    // token-major bf16
    float* c1   = (float*)alloc((size_t)B * C1 * NPB * 4);
    float* c2   = (float*)alloc((size_t)B * C1 * NPB * 4);
    float* scbf = (float*)alloc((size_t)B * C1 * NPB * 4);
    float* h2a  = (float*)alloc((size_t)B * C1 * NPB * 4);
    float* temb = (float*)alloc((size_t)B * C1 * 4);
    unsigned short* w1b  = (unsigned short*)alloc((size_t)256 * 1152 * 2);
    unsigned short* w2b  = (unsigned short*)alloc((size_t)256 * 2304 * 2);
    unsigned short* w3b  = (unsigned short*)alloc((size_t)256 * 2304 * 2);
    unsigned short* w4b  = (unsigned short*)alloc((size_t)256 * 2304 * 2);
    unsigned short* wscb = (unsigned short*)alloc((size_t)256 * 128 * 2);
    unsigned short* wqb  = (unsigned short*)alloc((size_t)32 * 256 * 2);
    unsigned short* wkb  = (unsigned short*)alloc((size_t)32 * 256 * 2);
    unsigned short* wvb  = (unsigned short*)alloc((size_t)256 * 256 * 2);
    unsigned short* q16  = (unsigned short*)alloc((size_t)NT * 32 * 2);
    unsigned short* k16  = (unsigned short*)alloc((size_t)NT * 32 * 2);
    unsigned short* v16  = (unsigned short*)alloc((size_t)NT * 256 * 2);

    float* outp  = (float*)d_out;                          // pooled [4][256][32][32]
    float* h1out = outp + (long long)B * C1 * 32 * 32;     // h1 [4][256][64][64]
    float* h2out = h1out + (long long)B * C1 * NPB;        // h2 [4][256][64][64]

    auto cgrid = [](long long n) { return dim3((unsigned)((n + 255) / 256)); };

    // weight casts
    cast_bf16_k<<<cgrid(256 * 1152), 256, 0, stream>>>(r1c1w, w1b, 256 * 1152);
    cast_bf16_k<<<cgrid(256 * 2304), 256, 0, stream>>>(r1c2w, w2b, 256 * 2304);
    cast_bf16_k<<<cgrid(256 * 2304), 256, 0, stream>>>(r2c1w, w3b, 256 * 2304);
    cast_bf16_k<<<cgrid(256 * 2304), 256, 0, stream>>>(r2c2w, w4b, 256 * 2304);
    cast_bf16_k<<<cgrid(256 * 128),  256, 0, stream>>>(r1scw, wscb, 256 * 128);
    cast_bf16_k<<<cgrid(32 * 256),   256, 0, stream>>>(aqw, wqb, 32 * 256);
    cast_bf16_k<<<cgrid(32 * 256),   256, 0, stream>>>(akw, wkb, 32 * 256);
    cast_bf16_k<<<cgrid(256 * 256),  256, 0, stream>>>(avw, wvb, 256 * 256);

    dim3 gemmBlk(256);
    dim3 gConv(8, 128);    // M=256 (32-row tiles), N=16384 (128-col tiles)
    dim3 gQK(1, 128);      // M=32

    // ---- residual block 1 ----
    im2col3x3_k<128><<<cgrid(NT * C0 * 9), 256, 0, stream>>>(x, ic);
    wmma_gemm_k<<<gConv, gemmBlk, 0, stream>>>(w1b, ic, nullptr, c1, 256, 1152, 0, 0);
    time_emb_k<<<dim3(B), dim3(256), 0, stream>>>(t, r1tw, r1tb, temb, 512, 256);
    group_norm_k<<<dim3(B * 256), dim3(256), 0, stream>>>(c1, r1g1g, r1g1b, temb, 1, 256);
    im2col3x3_k<256><<<cgrid(NT * C1 * 9), 256, 0, stream>>>(c1, ic);
    wmma_gemm_k<<<gConv, gemmBlk, 0, stream>>>(w2b, ic, nullptr, c2, 256, 2304, 0, 0);
    group_norm_k<<<dim3(B * 256), dim3(256), 0, stream>>>(c2, r1g2g, r1g2b, nullptr, 0, 256);
    transpose_cast_k<128><<<cgrid(NT * C0), 256, 0, stream>>>(x, xt);
    wmma_gemm_k<<<gConv, gemmBlk, 0, stream>>>(wscb, xt, nullptr, scbf, 256, 128, 0, 0);
    group_norm_k<<<dim3(B * 8), dim3(256), 0, stream>>>(scbf, r1scg, r1scb, nullptr, 0, 8);
    add_silu_k<<<cgrid((long long)B * C1 * NPB), 256, 0, stream>>>(c2, scbf, h1out, (long long)B * C1 * NPB);

    // ---- residual block 2 (identity shortcut) ----
    im2col3x3_k<256><<<cgrid(NT * C1 * 9), 256, 0, stream>>>(h1out, ic);
    wmma_gemm_k<<<gConv, gemmBlk, 0, stream>>>(w3b, ic, nullptr, c1, 256, 2304, 0, 0);
    time_emb_k<<<dim3(B), dim3(256), 0, stream>>>(t, r2tw, r2tb, temb, 512, 256);
    group_norm_k<<<dim3(B * 256), dim3(256), 0, stream>>>(c1, r2g1g, r2g1b, temb, 1, 256);
    im2col3x3_k<256><<<cgrid(NT * C1 * 9), 256, 0, stream>>>(c1, ic);
    wmma_gemm_k<<<gConv, gemmBlk, 0, stream>>>(w4b, ic, nullptr, c2, 256, 2304, 0, 0);
    group_norm_k<<<dim3(B * 256), dim3(256), 0, stream>>>(c2, r2g2g, r2g2b, nullptr, 0, 256);
    add_silu_k<<<cgrid((long long)B * C1 * NPB), 256, 0, stream>>>(c2, h1out, h2a, (long long)B * C1 * NPB);

    // ---- self-attention ----
    transpose_cast_k<256><<<cgrid(NT * C1), 256, 0, stream>>>(h2a, xt);
    wmma_gemm_k<<<gQK,  gemmBlk, 0, stream>>>(wqb, xt, aqb, q16, 32, 256, 1, 1);
    wmma_gemm_k<<<gQK,  gemmBlk, 0, stream>>>(wkb, xt, akb, k16, 32, 256, 1, 1);
    wmma_gemm_k<<<gConv, gemmBlk, 0, stream>>>(wvb, xt, avb, v16, 256, 256, 0, 1);
    flash_attn_k<<<dim3(B * (NPB / 16)), dim3(32), 0, stream>>>(q16, k16, v16, h2a, h2out, agam);

    // ---- max pool ----
    maxpool_k<<<cgrid((long long)B * C1 * 32 * 32), 256, 0, stream>>>(h2out, outp);
}
